// MultiHeadedAttention_18408229830765
// MI455X (gfx1250) — matrix operand order repaired and checked by reading, not compiled, not run on hardware
//
#include <hip/hip_runtime.h>
#include <hip/hip_bf16.h>

// MultiHeadedAttention for MI455X (gfx1250), bf16 WMMA path.
// B=2, S=2048, H=1024, NH=16, PH=64.  Workspace use: ~67 MB.

typedef __bf16 bf16_t;
typedef __attribute__((ext_vector_type(16))) __bf16 v16bf;
typedef __attribute__((ext_vector_type(8)))  __bf16 v8bf;
typedef __attribute__((ext_vector_type(8)))  float  v8f;

#define B_DIM  2
#define S_DIM  2048
#define H_DIM  1024
#define NH_DIM 16
#define PH_DIM 64
#define M_DIM  (B_DIM * S_DIM)   // 4096

// ---------------------------------------------------------------- fragments

__device__ __forceinline__ v16bf make_v16(v8bf lo, v8bf hi) {
    return __builtin_shufflevector(lo, hi, 0, 1, 2, 3, 4, 5, 6, 7,
                                           8, 9, 10, 11, 12, 13, 14, 15);
}

// A fragment, 16x32 bf16, row-major source with leading dim ldk.
// ISA layout: lanes 0-15 row M=L, k = {kk..kk+7, kk+16..kk+23};
//             lanes 16-31 row M=L-16, k shifted by +8.
__device__ __forceinline__ v16bf load_a_frag(const bf16_t* __restrict__ A,
                                             int row0, int kk, int ldk) {
    int lane = threadIdx.x & 31;
    int row  = row0 + (lane & 15);
    int kb   = kk + ((lane >> 4) << 3);
    const bf16_t* p = A + (size_t)row * ldk + kb;
    v8bf lo = *(const v8bf*)p;
    v8bf hi = *(const v8bf*)(p + 16);
    return make_v16(lo, hi);
}

// B fragment, 32x16 bf16, loaded from B^T stored row-major [N, K] (ldk).
// Lane holds column n = col0 + (L&15); k = {kb..kb+7, kb+16..kb+23},
// kb = kk + 8*(L>>4)  -> two 16-byte loads (matches A fragment).
__device__ __forceinline__ v16bf load_b_frag(const bf16_t* __restrict__ Bt,
                                             int col0, int kk, int ldk) {
    int lane = threadIdx.x & 31;
    int n    = col0 + (lane & 15);
    int kb   = kk + ((lane >> 4) << 3);
    const bf16_t* p = Bt + (size_t)n * ldk + kb;
    v8bf lo = *(const v8bf*)p;
    v8bf hi = *(const v8bf*)(p + 16);
    return make_v16(lo, hi);
}

__device__ __forceinline__ v8f wmma_bf16(v16bf a, v16bf b, v8f c) {
    // D = A x B + C, 16x16x32 bf16 -> f32
    return __builtin_amdgcn_wmma_f32_16x16x32_bf16(false, a, false, b,
                                                   (short)0, c, false, false);
}

__device__ __forceinline__ v8f v8f_zero() {
    return (v8f){0.f, 0.f, 0.f, 0.f, 0.f, 0.f, 0.f, 0.f};
}

// ---------------------------------------------------------------- convert

__global__ __launch_bounds__(256) void f32_to_bf16_kernel(
        const float* __restrict__ in, bf16_t* __restrict__ out, int n) {
    int i = blockIdx.x * blockDim.x + threadIdx.x;
    int stride = gridDim.x * blockDim.x;
    for (; i < n; i += stride) out[i] = (bf16_t)in[i];
}

// ---------------------------------------------------------------- GEMM
// C[M,N] = A[M,K](bf16) * Bt[N,K]^T(bf16) + bias[N]
// MODE 0: f32 out, plain row-major [M,N]
// MODE 1: bf16 out, head-split   [b, h, s, d]   (Q, K)
// MODE 2: bf16 out, head-split-T [b, h, d, s]   (V, pre-transposed for P*V)
// Block tile 128x256, 8 waves (2x4), wave tile 64x64 (4x4 WMMA tiles):
// 16 WMMA per k-step against 8 fragment loads.

template <int MODE>
__global__ __launch_bounds__(256) void gemm_wmma_kernel(
        const bf16_t* __restrict__ A, const bf16_t* __restrict__ Bt,
        const float* __restrict__ bias, void* __restrict__ outp,
        int Mdim, int Ndim, int Kdim) {
    int lane   = threadIdx.x & 31;
    int wave   = threadIdx.x >> 5;
    int wave_m = wave & 1;       // 2 waves along M
    int wave_n = wave >> 1;      // 4 waves along N
    int m0 = blockIdx.y * 128 + wave_m * 64;
    int n0 = blockIdx.x * 256 + wave_n * 64;

    v8f acc[4][4];
#pragma unroll
    for (int i = 0; i < 4; ++i)
#pragma unroll
        for (int j = 0; j < 4; ++j) acc[i][j] = v8f_zero();

    for (int kk = 0; kk < Kdim; kk += 32) {
        v16bf a[4], b[4];
#pragma unroll
        for (int i = 0; i < 4; ++i) a[i] = load_a_frag(A, m0 + 16 * i, kk, Kdim);
#pragma unroll
        for (int j = 0; j < 4; ++j) b[j] = load_b_frag(Bt, n0 + 16 * j, kk, Kdim);
#pragma unroll
        for (int i = 0; i < 4; ++i)
#pragma unroll
            for (int j = 0; j < 4; ++j)
                acc[i][j] = wmma_bf16(a[i], b[j], acc[i][j]);
    }

    int rbase = (lane >> 4) << 3;   // C layout: rows v + 8*(lane>=16)
#pragma unroll
    for (int i = 0; i < 4; ++i) {
#pragma unroll
        for (int j = 0; j < 4; ++j) {
            int ncol = n0 + 16 * j + (lane & 15);
            float bv = bias[ncol];
#pragma unroll
            for (int v = 0; v < 8; ++v) {
                int m = m0 + 16 * i + rbase + v;
                float val = acc[i][j][v] + bv;
                if (MODE == 0) {
                    ((float*)outp)[(size_t)m * Ndim + ncol] = val;
                } else {
                    int bb = m >> 11;              // m / S
                    int ss = m & (S_DIM - 1);
                    int hh = ncol >> 6;            // ncol / PH
                    int dd = ncol & (PH_DIM - 1);
                    if (MODE == 1)
                        ((bf16_t*)outp)[(((size_t)(bb * NH_DIM + hh)) * S_DIM + ss) * PH_DIM + dd] =
                            (bf16_t)val;
                    else
                        ((bf16_t*)outp)[(((size_t)(bb * NH_DIM + hh)) * PH_DIM + dd) * S_DIM + ss] =
                            (bf16_t)val;
                }
            }
        }
    }
}

// ---------------------------------------------------------------- attention
// One wave per 32-query tile (two 16-row WMMA tiles sharing K/V fragments).
// Online softmax over 32-key steps.
// Q,K: [b,h,s,d] bf16; Vt: [b,h,d,s] bf16; mask: [b,1,S,S] f32 (multiplicative,
// applied to scaled scores BEFORE softmax, per reference); ctx: [B*S, H] bf16.

__global__ __launch_bounds__(128) void attn_wmma_kernel(
        const bf16_t* __restrict__ Q, const bf16_t* __restrict__ Kh,
        const bf16_t* __restrict__ Vt, const float* __restrict__ mask,
        bf16_t* __restrict__ ctx) {
    __shared__ __align__(16) bf16_t plds[4][2][16 * 32];   // per-wave P staging

    int lane = threadIdx.x & 31;
    int wave = threadIdx.x >> 5;
    int tile = blockIdx.x * 4 + wave;
    const int QT = S_DIM / 32;                 // 64 query-pair tiles per (b,h)
    int b  = tile / (NH_DIM * QT);
    int r  = tile % (NH_DIM * QT);
    int h  = r / QT;
    int qb = r % QT;                           // 32-query tile index

    const bf16_t* qptr = Q  + (((size_t)(b * NH_DIM + h)) * S_DIM + (size_t)qb * 32) * PH_DIM;
    const bf16_t* kptr = Kh + ((size_t)(b * NH_DIM + h)) * S_DIM * PH_DIM;
    const bf16_t* vptr = Vt + ((size_t)(b * NH_DIM + h)) * PH_DIM * S_DIM;
    const float*  mptr = mask + (size_t)b * S_DIM * S_DIM + (size_t)(qb * 32) * S_DIM;

    // Q tiles: 2 x (16x64) as A fragments (d = 0..31, 32..63), reused all iters.
    v16bf aq[2][2];
#pragma unroll
    for (int u = 0; u < 2; ++u) {
        aq[u][0] = load_a_frag(qptr, 16 * u, 0,  PH_DIM);
        aq[u][1] = load_a_frag(qptr, 16 * u, 32, PH_DIM);
    }

    v8f acc[2][4];
    float m_i[2][8], l_i[2][8];
#pragma unroll
    for (int u = 0; u < 2; ++u) {
#pragma unroll
        for (int j = 0; j < 4; ++j) acc[u][j] = v8f_zero();
#pragma unroll
        for (int v = 0; v < 8; ++v) { m_i[u][v] = -3.0e38f; l_i[u][v] = 0.f; }
    }

    int rbase = (lane >> 4) << 3;

    for (int kb = 0; kb < S_DIM; kb += 32) {
        if (kb + 32 < S_DIM) {                          // global_prefetch_b8
            __builtin_prefetch(kptr + (size_t)(kb + 32) * PH_DIM, 0, 0);
            __builtin_prefetch(vptr + (kb + 32), 0, 0);
        }

        // ---- scores S[2][16 x 32] = (Q K^T) / 8 * mask ; K frags shared by u
        v8f sf[2][2];
#pragma unroll
        for (int t = 0; t < 2; ++t) {
            v16bf bk0 = load_b_frag(kptr, kb + 16 * t, 0,  PH_DIM);
            v16bf bk1 = load_b_frag(kptr, kb + 16 * t, 32, PH_DIM);
            int keycol = kb + 16 * t + (lane & 15);
#pragma unroll
            for (int u = 0; u < 2; ++u) {
                v8f c = v8f_zero();
                c = wmma_bf16(aq[u][0], bk0, c);
                c = wmma_bf16(aq[u][1], bk1, c);
#pragma unroll
                for (int v = 0; v < 8; ++v) {
                    float mval = mptr[(size_t)(16 * u + rbase + v) * S_DIM + keycol];
                    c[v] = c[v] * 0.125f * mval;
                }
                sf[u][t] = c;
            }
        }

        // ---- online softmax update (rows live in 16-lane halves)
#pragma unroll
        for (int u = 0; u < 2; ++u) {
            bf16_t* myp = plds[wave][u];
#pragma unroll
            for (int v = 0; v < 8; ++v) {
                float x = fmaxf(sf[u][0][v], sf[u][1][v]);
#pragma unroll
                for (int o = 1; o < 16; o <<= 1) x = fmaxf(x, __shfl_xor(x, o, 32));
                float mn    = fmaxf(m_i[u][v], x);
                float alpha = __expf(m_i[u][v] - mn);
                float p0    = __expf(sf[u][0][v] - mn);
                float p1    = __expf(sf[u][1][v] - mn);
                float ps    = p0 + p1;
#pragma unroll
                for (int o = 1; o < 16; o <<= 1) ps += __shfl_xor(ps, o, 32);
                l_i[u][v] = l_i[u][v] * alpha + ps;
                m_i[u][v] = mn;
#pragma unroll
                for (int j = 0; j < 4; ++j) acc[u][j][v] *= alpha;
                int row = rbase + v;                    // C-layout -> LDS row-major
                myp[row * 32 + (lane & 15)]      = (bf16_t)p0;
                myp[row * 32 + 16 + (lane & 15)] = (bf16_t)p1;
            }
        }

        // ---- P (2 x 16x32) back as A fragments; O += P * V_tile, V shared by u
        {
            int prow = lane & 15;
            int c0   = (lane >> 4) << 3;
            v16bf pa[2];
#pragma unroll
            for (int u = 0; u < 2; ++u) {
                const bf16_t* myp = plds[wave][u];
                v8bf lo = *(const v8bf*)&myp[prow * 32 + c0];
                v8bf hi = *(const v8bf*)&myp[prow * 32 + 16 + c0];
                pa[u] = make_v16(lo, hi);
            }
#pragma unroll
            for (int j = 0; j < 4; ++j) {
                v16bf bvf = load_b_frag(vptr, j * 16, kb, S_DIM);  // Vt contiguous
#pragma unroll
                for (int u = 0; u < 2; ++u)
                    acc[u][j] = wmma_bf16(pa[u], bvf, acc[u][j]);
            }
        }
    }

    // ---- normalize and store merged-head ctx [B*S, H] bf16
#pragma unroll
    for (int u = 0; u < 2; ++u) {
        size_t mrow0 = (size_t)(b * S_DIM + qb * 32 + 16 * u);
#pragma unroll
        for (int v = 0; v < 8; ++v) {
            float inv  = 1.0f / l_i[u][v];
            size_t row = mrow0 + rbase + v;
#pragma unroll
            for (int j = 0; j < 4; ++j) {
                int col = h * PH_DIM + j * 16 + (lane & 15);
                ctx[row * H_DIM + col] = (bf16_t)(acc[u][j][v] * inv);
            }
        }
    }
}

// ---------------------------------------------------------------- launcher

extern "C" void kernel_launch(void* const* d_in, const int* in_sizes, int n_in,
                              void* d_out, int out_size, void* d_ws, size_t ws_size,
                              hipStream_t stream) {
    const float* key   = (const float*)d_in[0];
    const float* value = (const float*)d_in[1];
    const float* query = (const float*)d_in[2];
    const float* mask  = (const float*)d_in[3];
    /* d_in[4] = i (dropout seed index, unused: p=0) */
    const float* Wq = (const float*)d_in[5];
    const float* bq = (const float*)d_in[6];
    const float* Wk = (const float*)d_in[7];
    const float* bk = (const float*)d_in[8];
    const float* Wv = (const float*)d_in[9];
    const float* bv = (const float*)d_in[10];
    const float* Wo = (const float*)d_in[11];
    const float* bo = (const float*)d_in[12];
    float* out = (float*)d_out;

    const size_t X = (size_t)M_DIM * H_DIM;   // 4194304 activation elements
    const size_t W = (size_t)H_DIM * H_DIM;   // 1048576 weight elements
    bf16_t* ws  = (bf16_t*)d_ws;              // (7X + 4W) * 2B ~= 67 MB
    bf16_t* xq  = ws;
    bf16_t* xk  = xq + X;
    bf16_t* xv  = xk + X;
    bf16_t* wq  = xv + X;
    bf16_t* wk  = wq + W;
    bf16_t* wv  = wk + W;
    bf16_t* wo  = wv + W;
    bf16_t* qh  = wo + W;     // [b,h,s,d]
    bf16_t* kh  = qh + X;     // [b,h,s,d]
    bf16_t* vt  = kh + X;     // [b,h,d,s]
    bf16_t* ctx = vt + X;     // [B*S, H]

    // 1) downconvert activations + weights to bf16
    f32_to_bf16_kernel<<<1024, 256, 0, stream>>>(query, xq, (int)X);
    f32_to_bf16_kernel<<<1024, 256, 0, stream>>>(key,   xk, (int)X);
    f32_to_bf16_kernel<<<1024, 256, 0, stream>>>(value, xv, (int)X);
    f32_to_bf16_kernel<<<512,  256, 0, stream>>>(Wq, wq, (int)W);
    f32_to_bf16_kernel<<<512,  256, 0, stream>>>(Wk, wk, (int)W);
    f32_to_bf16_kernel<<<512,  256, 0, stream>>>(Wv, wv, (int)W);
    f32_to_bf16_kernel<<<512,  256, 0, stream>>>(Wo, wo, (int)W);

    // 2) projections (x @ W^T + b) -> head-split layouts
    dim3 gblk(256), ggrd(H_DIM / 256, M_DIM / 128);   // (4, 32)
    gemm_wmma_kernel<1><<<ggrd, gblk, 0, stream>>>(xq, wq, bq, qh, M_DIM, H_DIM, H_DIM);
    gemm_wmma_kernel<1><<<ggrd, gblk, 0, stream>>>(xk, wk, bk, kh, M_DIM, H_DIM, H_DIM);
    gemm_wmma_kernel<2><<<ggrd, gblk, 0, stream>>>(xv, wv, bv, vt, M_DIM, H_DIM, H_DIM);

    // 3) masked softmax attention (flash-style), one wave per 32-query tile
    int tiles  = B_DIM * NH_DIM * (S_DIM / 32);       // 2048
    attn_wmma_kernel<<<tiles / 4, 128, 0, stream>>>(qh, kh, vt, mask, ctx);

    // 4) output projection: out = ctx @ Wo^T + bo   (f32)
    gemm_wmma_kernel<0><<<ggrd, gblk, 0, stream>>>(ctx, wo, bo, (void*)out,
                                                   M_DIM, H_DIM, H_DIM);
}
